// ImageAttention_23072564314558
// MI455X (gfx1250) — compile-verified, matrix operand order checked by reading, not run on hardware
//
#include <hip/hip_runtime.h>

typedef __bf16 bf16;
typedef __bf16 v16bf __attribute__((ext_vector_type(16)));
typedef float  v8f   __attribute__((ext_vector_type(8)));

#define P_TOK 4096
#define DMODEL 768
#define NH 12
#define HD 64

// K index held by VGPR-pair v (0..7) for the 16-bit A/B wave32 WMMA layout.
// lanes 0-15 (half=0): K = {0,1},{2,3},{4,5},{6,7},{16,17},{18,19},{20,21},{22,23}
// lanes 16-31 (half=1): +8
__device__ __forceinline__ int kidx(int v, int half) {
    return ((v < 4) ? (2 * v) : (16 + 2 * (v - 4))) + 8 * half;
}

// ---------------------------------------------------------------------------
// Kernel 1: qkv = x @ w_qkv^T + b_qkv  -> per-head bf16 Q (unscaled), K (pre-
// scaled by hd^-0.5), V.  Tiled 64x64, 4 waves, WMMA bf16 16x16x32.
// ---------------------------------------------------------------------------
__global__ __launch_bounds__(128)
void qkv_gemm_kernel(const float* __restrict__ x, const float* __restrict__ w,
                     const float* __restrict__ bias,
                     bf16* __restrict__ qb, bf16* __restrict__ kb,
                     bf16* __restrict__ vb)
{
    __shared__ bf16 As[64][34];
    __shared__ bf16 Bs[64][34];
    const int tM = blockIdx.x;            // 64 tiles of M=4096
    const int tN = blockIdx.y;            // 36 tiles of N=2304
    const int tid = threadIdx.x;
    const int wid = tid >> 5, lane = tid & 31;
    const int n = lane & 15, half = lane >> 4;

    v8f acc[4] = {};

    for (int kk = 0; kk < DMODEL; kk += 32) {
        {
            int row = tid >> 1;
            int c0 = (tid & 1) * 16;
            const float* xs = x + (size_t)(tM * 64 + row) * DMODEL + kk + c0;
            const float* ws = w + (size_t)(tN * 64 + row) * DMODEL + kk + c0;
            #pragma unroll
            for (int u = 0; u < 16; ++u) {
                As[row][c0 + u] = (bf16)xs[u];
                Bs[row][c0 + u] = (bf16)ws[u];
            }
        }
        __syncthreads();

        v16bf a;
        #pragma unroll
        for (int v = 0; v < 8; ++v) {
            int c = kidx(v, half);
            a[2 * v]     = As[16 * wid + n][c];
            a[2 * v + 1] = As[16 * wid + n][c + 1];
        }
        #pragma unroll
        for (int nb = 0; nb < 4; ++nb) {
            v16bf b;
            #pragma unroll
            for (int v = 0; v < 8; ++v) {
                int c = kidx(v, half);
                b[2 * v]     = Bs[nb * 16 + n][c];
                b[2 * v + 1] = Bs[nb * 16 + n][c + 1];
            }
            acc[nb] = __builtin_amdgcn_wmma_f32_16x16x32_bf16(
                false, a, false, b, (short)0, acc[nb], false, false);
        }
        __syncthreads();
    }

    const int i0 = tM * 64 + 16 * wid + 8 * half;
    #pragma unroll
    for (int nb = 0; nb < 4; ++nb) {
        int j = tN * 64 + nb * 16 + n;         // column in [0,2304)
        int part = j / 768;                    // 0:q 1:k 2:v (constant per tile)
        int hj = (j % 768) >> 6;
        int cj = j & 63;
        float bb = bias[j];
        #pragma unroll
        for (int r = 0; r < 8; ++r) {
            float val = acc[nb][r] + bb;
            size_t off = ((size_t)hj * P_TOK + (i0 + r)) * HD + cj;
            if (part == 0)      qb[off] = (bf16)val;
            else if (part == 1) kb[off] = (bf16)(val * 0.125f);  // hd^-0.5 folded
            else                vb[off] = (bf16)val;
        }
    }
}

// ---------------------------------------------------------------------------
// Kernel 2: decomposed rel-pos bias tables.
// rel_h[head][i][k] = sum_c q[head][i][c] * rel_pos_h[h1-k+63][c]  (q UNscaled)
// ---------------------------------------------------------------------------
__global__ __launch_bounds__(64)
void relbias_kernel(const bf16* __restrict__ qb,
                    const float* __restrict__ rph, const float* __restrict__ rpw,
                    float* __restrict__ relh, float* __restrict__ relw)
{
    __shared__ float qrow[HD];
    int bid = blockIdx.x;                 // head*4096 + i
    int head = bid >> 12;
    int i = bid & 4095;
    int h1 = i >> 6, w1 = i & 63;
    int k = threadIdx.x;                  // 64 threads, one k each
    qrow[k] = (float)qb[((size_t)head * P_TOK + i) * HD + k];
    __syncthreads();
    const float* ph = rph + (size_t)(h1 - k + 63) * HD;
    const float* pw = rpw + (size_t)(w1 - k + 63) * HD;
    float sh = 0.f, sw = 0.f;
    #pragma unroll 8
    for (int c = 0; c < HD; ++c) {
        float qc = qrow[c];
        sh += qc * ph[c];
        sw += qc * pw[c];
    }
    relh[((size_t)head * P_TOK + i) * HD + k] = sh;
    relw[((size_t)head * P_TOK + i) * HD + k] = sw;
}

// ---------------------------------------------------------------------------
// Async global->LDS staging helpers (gfx1250 ASYNCcnt path).
// ---------------------------------------------------------------------------
__device__ __forceinline__ void stage_chunk_async(bf16 (*__restrict__ Kdst)[72],
                                                  bf16 (*__restrict__ Vdst)[72],
                                                  const bf16* __restrict__ Kg,
                                                  const bf16* __restrict__ Vg, int tid)
{
    // 64x64 bf16 chunk = 512 B128 segments; thread handles 4 for K + 4 for V.
    #pragma unroll
    for (int u = 0; u < 4; ++u) {
        int s = tid + 128 * u;
        int jl = s >> 3, c = (s & 7) * 8;           // row, 8-elem column group
        unsigned kl = (unsigned)(uintptr_t)&Kdst[jl][c];   // low 32 = LDS addr
        unsigned vl = (unsigned)(uintptr_t)&Vdst[jl][c];
        const bf16* kg = Kg + (size_t)jl * HD + c;
        const bf16* vg = Vg + (size_t)jl * HD + c;
        asm volatile("global_load_async_to_lds_b128 %0, %1, off"
                     :: "v"(kl), "v"(kg) : "memory");
        asm volatile("global_load_async_to_lds_b128 %0, %1, off"
                     :: "v"(vl), "v"(vg) : "memory");
    }
}

// ---------------------------------------------------------------------------
// Kernel 3: flash attention per (head, 64-query tile); 4 waves x 16 rows.
// 4096 keys streamed in 64-key chunks, double-buffered async LDS staging,
// online softmax, bf16 WMMA 16x16x32.
// ---------------------------------------------------------------------------
__global__ __launch_bounds__(128)
void flash_attn_kernel(const bf16* __restrict__ qb, const bf16* __restrict__ kb,
                       const bf16* __restrict__ vb,
                       const float* __restrict__ relh, const float* __restrict__ relw,
                       float* __restrict__ obuf)
{
    __shared__ __align__(16) bf16 Ks[2][64][72];  // K chunk [key][c], 144B rows
    __shared__ __align__(16) bf16 Vs[2][64][72];  // V chunk [key][c]
    __shared__ float RH[64][65];                  // rel_h rows for this q tile
    __shared__ bf16  Pb[4][16][66];               // per-wave P transpose bounce

    const int head = blockIdx.y;
    const int q0 = blockIdx.x * 64;
    const int tid = threadIdx.x, wid = tid >> 5, lane = tid & 31;
    const int n = lane & 15, half = lane >> 4;
    const int lmBase = 16 * wid + 8 * half;

    const bf16* Qh = qb + (size_t)head * P_TOK * HD;
    const bf16* Kh = kb + (size_t)head * P_TOK * HD;
    const bf16* Vh = vb + (size_t)head * P_TOK * HD;

    // prologue: kick off async staging of chunk 0 immediately
    stage_chunk_async(Ks[0], Vs[0], Kh, Vh, tid);

    // stage rel_h rows (64 x 64 f32) into LDS
    for (int u = tid; u < 64 * 64; u += 128) {
        int r = u >> 6, c = u & 63;
        RH[r][c] = relh[((size_t)head * P_TOK + q0 + r) * HD + c];
    }

    // rel_w values this lane will ever need: rows lmBase+r, cols t*16+n.
    float rwreg[8][4];
    #pragma unroll
    for (int r = 0; r < 8; ++r)
        #pragma unroll
        for (int t = 0; t < 4; ++t)
            rwreg[r][t] = relw[((size_t)head * P_TOK + q0 + lmBase + r) * HD + t * 16 + n];

    // Q A-fragments, held for the whole kernel (rows 16*wid + (lane&15))
    const size_t qrowA = (size_t)(q0 + 16 * wid + n) * HD;
    v16bf qa0, qa1;
    #pragma unroll
    for (int v = 0; v < 8; ++v) {
        int c = kidx(v, half);
        qa0[2 * v]     = Qh[qrowA + c];
        qa0[2 * v + 1] = Qh[qrowA + c + 1];
        qa1[2 * v]     = Qh[qrowA + 32 + c];
        qa1[2 * v + 1] = Qh[qrowA + 32 + c + 1];
    }

    v8f acc[4] = {};
    float mrow[8], lrow[8];
    #pragma unroll
    for (int r = 0; r < 8; ++r) { mrow[r] = -3.0e38f; lrow[r] = 0.f; }

    for (int it = 0; it < 64; ++it) {
        const int cur = it & 1;
        if (it + 1 < 64) {
            // safe: buffer (it+1)&1 was last read in iteration it-1, whose
            // end-of-iteration barrier every wave has passed by now.
            stage_chunk_async(Ks[cur ^ 1], Vs[cur ^ 1],
                              Kh + (size_t)(it + 1) * 64 * HD,
                              Vh + (size_t)(it + 1) * 64 * HD, tid);
            asm volatile("s_wait_asynccnt 8" ::: "memory");   // chunk it done (us)
        } else {
            asm volatile("s_wait_asynccnt 0" ::: "memory");
        }
        __syncthreads();   // chunk it resident from all waves (incl. RH on it==0)

        // ---- S = Q * K^T : four 16x16 N-tiles over 64 keys ----
        v8f s4[4];
        #pragma unroll
        for (int t = 0; t < 4; ++t) {
            v16bf b0, b1;
            #pragma unroll
            for (int v = 0; v < 8; ++v) {
                int c = kidx(v, half);
                b0[2 * v]     = Ks[cur][t * 16 + n][c];
                b0[2 * v + 1] = Ks[cur][t * 16 + n][c + 1];
                b1[2 * v]     = Ks[cur][t * 16 + n][32 + c];
                b1[2 * v + 1] = Ks[cur][t * 16 + n][32 + c + 1];
            }
            v8f z = {};
            z     = __builtin_amdgcn_wmma_f32_16x16x32_bf16(false, qa0, false, b0, (short)0, z, false, false);
            s4[t] = __builtin_amdgcn_wmma_f32_16x16x32_bf16(false, qa1, false, b1, (short)0, z, false, false);
        }

        // ---- decomposed rel-pos bias + online softmax (h2 == it, w2 == t*16+n)
        float newmax[8];
        #pragma unroll
        for (int r = 0; r < 8; ++r) {
            float bh = RH[lmBase + r][it];
            float v0 = s4[0][r] + bh + rwreg[r][0];
            float v1 = s4[1][r] + bh + rwreg[r][1];
            float v2 = s4[2][r] + bh + rwreg[r][2];
            float v3 = s4[3][r] + bh + rwreg[r][3];
            s4[0][r] = v0; s4[1][r] = v1; s4[2][r] = v2; s4[3][r] = v3;
            float mx = fmaxf(fmaxf(v0, v1), fmaxf(v2, v3));
            mx = fmaxf(mx, __shfl_xor(mx, 1, 32));
            mx = fmaxf(mx, __shfl_xor(mx, 2, 32));
            mx = fmaxf(mx, __shfl_xor(mx, 4, 32));
            mx = fmaxf(mx, __shfl_xor(mx, 8, 32));
            newmax[r] = fmaxf(mrow[r], mx);
        }
        #pragma unroll
        for (int r = 0; r < 8; ++r) {
            int m = 8 * half + r;
            float e0 = __expf(s4[0][r] - newmax[r]);
            float e1 = __expf(s4[1][r] - newmax[r]);
            float e2 = __expf(s4[2][r] - newmax[r]);
            float e3 = __expf(s4[3][r] - newmax[r]);
            Pb[wid][m][n]      = (bf16)e0;
            Pb[wid][m][16 + n] = (bf16)e1;
            Pb[wid][m][32 + n] = (bf16)e2;
            Pb[wid][m][48 + n] = (bf16)e3;
            float rs = (e0 + e1) + (e2 + e3);
            rs += __shfl_xor(rs, 1, 32);
            rs += __shfl_xor(rs, 2, 32);
            rs += __shfl_xor(rs, 4, 32);
            rs += __shfl_xor(rs, 8, 32);
            float fac = __expf(mrow[r] - newmax[r]);
            lrow[r] = lrow[r] * fac + rs;
            mrow[r] = newmax[r];
            #pragma unroll
            for (int nb = 0; nb < 4; ++nb) acc[nb][r] *= fac;
        }

        // ---- P: C-layout -> A-layout via per-wave LDS bounce ----
        asm volatile("s_wait_dscnt 0" ::: "memory");  // cross-lane DS RAW in-wave
        v16bf pa0, pa1;
        #pragma unroll
        for (int v = 0; v < 8; ++v) {
            int c = kidx(v, half);
            pa0[2 * v]     = Pb[wid][n][c];
            pa0[2 * v + 1] = Pb[wid][n][c + 1];
            pa1[2 * v]     = Pb[wid][n][32 + c];
            pa1[2 * v + 1] = Pb[wid][n][32 + c + 1];
        }

        // ---- O += P * V  (V row-major: b[k][col] = Vs[key][col]) ----
        #pragma unroll
        for (int nb = 0; nb < 4; ++nb) {
            v16bf b0, b1;
            #pragma unroll
            for (int v = 0; v < 8; ++v) {
                int c = kidx(v, half);              // key index pairs
                b0[2 * v]     = Vs[cur][c][nb * 16 + n];
                b0[2 * v + 1] = Vs[cur][c + 1][nb * 16 + n];
                b1[2 * v]     = Vs[cur][32 + c][nb * 16 + n];
                b1[2 * v + 1] = Vs[cur][32 + c + 1][nb * 16 + n];
            }
            acc[nb] = __builtin_amdgcn_wmma_f32_16x16x32_bf16(
                false, pa0, false, b0, (short)0, acc[nb], false, false);
            acc[nb] = __builtin_amdgcn_wmma_f32_16x16x32_bf16(
                false, pa1, false, b1, (short)0, acc[nb], false, false);
        }
        __syncthreads();   // all reads of buffer `cur` done before reuse
    }

    // ---- normalize + write per-head output (f32, [head][pos][64]) ----
    float* Oh = obuf + (size_t)head * P_TOK * HD;
    #pragma unroll
    for (int r = 0; r < 8; ++r) {
        int qi = q0 + lmBase + r;
        float inv = 1.f / lrow[r];
        #pragma unroll
        for (int nb = 0; nb < 4; ++nb)
            Oh[(size_t)qi * HD + nb * 16 + n] = acc[nb][r] * inv;
    }
}

// ---------------------------------------------------------------------------
// Kernel 4: out = O @ w_proj^T + b_proj, gathering head-major O.
// ---------------------------------------------------------------------------
__global__ __launch_bounds__(128)
void proj_gemm_kernel(const float* __restrict__ obuf, const float* __restrict__ w,
                      const float* __restrict__ bias, float* __restrict__ out)
{
    __shared__ bf16 As[64][34];
    __shared__ bf16 Bs[64][34];
    const int tM = blockIdx.x;            // 64
    const int tN = blockIdx.y;            // 12
    const int tid = threadIdx.x;
    const int wid = tid >> 5, lane = tid & 31;
    const int n = lane & 15, half = lane >> 4;

    v8f acc[4] = {};

    for (int kk = 0; kk < DMODEL; kk += 32) {
        {
            int row = tid >> 1;
            int c0 = (tid & 1) * 16;
            int kg = kk + c0;                       // 16-aligned -> single head
            const float* osrc = obuf + ((size_t)(kg >> 6) * P_TOK + tM * 64 + row) * HD + (kg & 63);
            const float* ws = w + (size_t)(tN * 64 + row) * DMODEL + kg;
            #pragma unroll
            for (int u = 0; u < 16; ++u) {
                As[row][c0 + u] = (bf16)osrc[u];
                Bs[row][c0 + u] = (bf16)ws[u];
            }
        }
        __syncthreads();

        v16bf a;
        #pragma unroll
        for (int v = 0; v < 8; ++v) {
            int c = kidx(v, half);
            a[2 * v]     = As[16 * wid + n][c];
            a[2 * v + 1] = As[16 * wid + n][c + 1];
        }
        #pragma unroll
        for (int nb = 0; nb < 4; ++nb) {
            v16bf b;
            #pragma unroll
            for (int v = 0; v < 8; ++v) {
                int c = kidx(v, half);
                b[2 * v]     = Bs[nb * 16 + n][c];
                b[2 * v + 1] = Bs[nb * 16 + n][c + 1];
            }
            acc[nb] = __builtin_amdgcn_wmma_f32_16x16x32_bf16(
                false, a, false, b, (short)0, acc[nb], false, false);
        }
        __syncthreads();
    }

    const int i0 = tM * 64 + 16 * wid + 8 * half;
    #pragma unroll
    for (int nb = 0; nb < 4; ++nb) {
        int j = tN * 64 + nb * 16 + n;
        float bb = bias[j];
        #pragma unroll
        for (int r = 0; r < 8; ++r)
            out[(size_t)(i0 + r) * DMODEL + j] = acc[nb][r] + bb;
    }
}

// ---------------------------------------------------------------------------
extern "C" void kernel_launch(void* const* d_in, const int* in_sizes, int n_in,
                              void* d_out, int out_size, void* d_ws, size_t ws_size,
                              hipStream_t stream)
{
    const float* x      = (const float*)d_in[0];
    const float* w_qkv  = (const float*)d_in[1];
    const float* b_qkv  = (const float*)d_in[2];
    const float* w_proj = (const float*)d_in[3];
    const float* b_proj = (const float*)d_in[4];
    const float* rph    = (const float*)d_in[5];
    const float* rpw    = (const float*)d_in[6];

    char* ws = (char*)d_ws;
    // per-head bf16 Q/K/V: 12*4096*64*2B = 6,291,456 each
    bf16*  qb = (bf16*)(ws);
    bf16*  kb = (bf16*)(ws + 6291456);
    bf16*  vb = (bf16*)(ws + 12582912);
    float* ob = (float*)(ws + 18874368);   // 12*4096*64 f32 = 12,582,912 B
    float* rh = (float*)(ws + 31457280);   // 12,582,912 B
    float* rw = (float*)(ws + 44040192);   // ends at 56,623,104 B

    qkv_gemm_kernel<<<dim3(64, 36), 128, 0, stream>>>(x, w_qkv, b_qkv, qb, kb, vb);
    relbias_kernel<<<dim3(NH * P_TOK), 64, 0, stream>>>(qb, rph, rpw, rh, rw);
    flash_attn_kernel<<<dim3(64, NH), 128, 0, stream>>>(qb, kb, vb, rh, rw, ob);
    proj_gemm_kernel<<<dim3(64, NH), 128, 0, stream>>>(ob, w_proj, b_proj, (float*)d_out);
}
